// CustomLSTM_82600811036952
// MI455X (gfx1250) — matrix operand order repaired and checked by reading, not compiled, not run on hardware
//
#include <hip/hip_runtime.h>
#include <hip/hip_bf16.h>

// LSTM: B=64, S=512, D=H=1024.
//  - xg = bf16(x @ Wi^T + bi) as one big bf16-WMMA GEMM (ws-resident)
//  - 512 sequential step kernels: gates = xg_t + h @ Wh^T + bh fused with the
//    LSTM cell update (block owns cols j..j+15 of ALL four gates).
//  - Matmuls: v_wmma_f32_16x16x32_bf16 (f32 accumulate).
//  - Step-kernel tiles (already bf16) move via the Tensor Data Mover with
//    double-buffered LDS: chunk i+1's tensor_load_to_lds is issued before
//    s_wait_tensorcnt drains chunk i, overlapping DMA with WMMA on the
//    latency-critical recurrent path.

#define Bsz 64
#define Ssz 512
#define Dsz 1024
#define Hsz 1024
#define G4  (4 * Hsz)

typedef __bf16 bf16_t;
typedef __attribute__((ext_vector_type(16))) bf16_t bf16x16;
typedef __attribute__((ext_vector_type(8)))  float  f32x8;
typedef __attribute__((ext_vector_type(4)))  unsigned int u32x4;
typedef __attribute__((ext_vector_type(8)))  int i32x8;
typedef __attribute__((ext_vector_type(4)))  int i32x4;

struct __align__(16) U4  { unsigned int a, b, c, d; };
struct __align__(16) F4  { float x, y, z, w; };
struct __align__(8)  US4 { unsigned short a, b, c, d; };

union Frag { bf16x16 v; U4 q[2]; };

__device__ __forceinline__ unsigned short f2bf(float f) {
  bf16_t h = (bf16_t)f;                        // hardware v_cvt_*bf16_f32
  return __builtin_bit_cast(unsigned short, h);
}
__device__ __forceinline__ float bf2f(unsigned short s) {
  return __uint_as_float(((unsigned int)s) << 16);
}

#if __has_builtin(__builtin_amdgcn_tensor_load_to_lds) && \
    __has_builtin(__builtin_amdgcn_s_wait_tensorcnt)
#define USE_TDM 1
#else
#define USE_TDM 0
#endif

#if USE_TDM
// Build a D# (ISA 8.3-8.6) and issue TENSOR_LOAD_TO_LDS (6-arg clang-23 form:
// groups 0-3 = VADDR0..VADDR3, 5th vector = VADDR4, unused for tensor ops ->
// zero-filled, then cache-policy immediate).
// Tile: tile_d0 elems (bf16) x tile_d1 (stride d0_stride elems) x tile_d2
// (stride d1_stride elems).  pad_on: +4 DWORDs LDS pad per 16 DWORDs stored
// (=> 40-short row stride for tile_d0==32).
__device__ __forceinline__ void tdm_load(
    unsigned lds, const void* base,
    unsigned tile_d0, unsigned tile_d1, unsigned tile_d2,
    unsigned long long d0_stride, unsigned long long d1_stride, int pad_on) {
  unsigned long long ga = (unsigned long long)(size_t)base;
  u32x4 g0 = {0u, 0u, 0u, 0u};
  g0[0] = 1u;                                           // count=1 valid user D#
  g0[1] = lds;                                          // lds_addr (bytes)
  g0[2] = (unsigned)ga;                                 // global_addr[31:0]
  g0[3] = (unsigned)((ga >> 32) & 0x01FFFFFFu) | (2u << 30);  // addr[56:32], type=2
  i32x8 g1 = {0, 0, 0, 0, 0, 0, 0, 0};
  unsigned w0 = (1u << 16);                             // data_size = 2 bytes
  if (pad_on) w0 |= (1u << 20) | (3u << 22) | (3u << 25);  // pad: 4dw per 16dw
  g1[0] = (int)w0;
  g1[1] = (int)((tile_d0 & 0xFFFFu) << 16);             // tensor_dim0 = tile_d0
  g1[2] = (int)(((tile_d0 >> 16) & 0xFFFFu) | ((tile_d1 & 0xFFFFu) << 16)); // tensor_dim1
  g1[3] = (int)(((tile_d1 >> 16) & 0xFFFFu) | ((tile_d0 & 0xFFFFu) << 16)); // tile_dim0
  g1[4] = (int)((tile_d1 & 0xFFFFu) | ((tile_d2 & 0xFFFFu) << 16));         // tile_dim1/2
  g1[5] = (int)(unsigned)(d0_stride & 0xFFFFFFFFull);   // dim0_stride[31:0]
  g1[6] = (int)((unsigned)((d0_stride >> 32) & 0xFFFFull) |
                (unsigned)((d1_stride & 0xFFFFull) << 16));
  g1[7] = (int)(unsigned)((d1_stride >> 16) & 0xFFFFFFFFull);
  i32x4 g2 = {(int)(tile_d2 ? tile_d2 : 1u), 0, 0, 0};  // tensor_dim2
  i32x4 g3 = {0, 0, 0, 0};
  i32x8 g4 = {0, 0, 0, 0, 0, 0, 0, 0};                  // VADDR4: unused
  __builtin_amdgcn_tensor_load_to_lds(g0, g1, g2, g3, g4, 0);
}
#endif

// ---------------------------------------------------------------------------
// prep: Wh -> bf16, broadcast h0/c0 into per-batch state
// ---------------------------------------------------------------------------
__global__ __launch_bounds__(256) void lstm_prep(
    const float* __restrict__ Wh, const float* __restrict__ h0,
    const float* __restrict__ c0, unsigned short* __restrict__ whb,
    unsigned short* __restrict__ hb0, float* __restrict__ cbuf) {
  int i = blockIdx.x * 256 + threadIdx.x;
  if (i < G4 * Hsz) whb[i] = f2bf(Wh[i]);
  if (i < Bsz * Hsz) {
    hb0[i]  = f2bf(h0[i & (Hsz - 1)]);
    cbuf[i] = c0[i & (Hsz - 1)];
  }
}

// ---------------------------------------------------------------------------
// proj: xg[32768,4096] = bf16(x[32768,1024] @ Wi^T + bi)
// Block 128(M) x 128(N), K-chunk 32, 8 waves (wave w -> M row-tile w, 8
// N-tiles each => 8 wmma per A-fragment).  LDS rows padded to 40 shorts.
// ---------------------------------------------------------------------------
__global__ __launch_bounds__(256) void lstm_proj_gemm(
    const float* __restrict__ x, const float* __restrict__ Wi,
    const float* __restrict__ bi, unsigned short* __restrict__ xg) {
  __shared__ unsigned short sA[128 * 40];
  __shared__ unsigned short sB[128 * 40];
  const int tid  = threadIdx.x;
  const int wave = tid >> 5, lane = tid & 31;
  const int m0 = blockIdx.x * 128;
  const int n0 = blockIdx.y * 128;
  f32x8 acc[8] = {};

  for (int k0 = 0; k0 < Dsz; k0 += 32) {
    __syncthreads();
#pragma unroll
    for (int i = 0; i < 4; ++i) {           // A: 128 rows x 32 K (f32->bf16)
      int idx = tid + i * 256;
      int row = idx >> 3, c4 = idx & 7;
      F4 f = *(const F4*)(x + (size_t)(m0 + row) * Dsz + k0 + c4 * 4);
      US4 u; u.a = f2bf(f.x); u.b = f2bf(f.y); u.c = f2bf(f.z); u.d = f2bf(f.w);
      *(US4*)&sA[row * 40 + c4 * 4] = u;
    }
#pragma unroll
    for (int i = 0; i < 4; ++i) {           // B: 128 N-rows x 32 K (Wi rows)
      int idx = tid + i * 256;
      int row = idx >> 3, c4 = idx & 7;
      F4 f = *(const F4*)(Wi + (size_t)(n0 + row) * Dsz + k0 + c4 * 4);
      US4 u; u.a = f2bf(f.x); u.b = f2bf(f.y); u.c = f2bf(f.z); u.d = f2bf(f.w);
      *(US4*)&sB[row * 40 + c4 * 4] = u;
    }
    __syncthreads();

    const int arow  = wave * 16 + (lane & 15);
    const int s     = (lane >> 4) * 8;      // unused; A also uses khalf: K 0-15 vs 16-31
    const int khalf = (lane >> 4) * 16;     // B: K 0-15 vs 16-31
    Frag a;
    a.q[0] = *(const U4*)&sA[arow * 40 + khalf];
    a.q[1] = *(const U4*)&sA[arow * 40 + khalf + 8];
#pragma unroll
    for (int nt = 0; nt < 8; ++nt) {
      const int brow = nt * 16 + (lane & 15);
      Frag b;
      b.q[0] = *(const U4*)&sB[brow * 40 + khalf];
      b.q[1] = *(const U4*)&sB[brow * 40 + khalf + 8];
      acc[nt] = __builtin_amdgcn_wmma_f32_16x16x32_bf16(
          false, a.v, false, b.v, (short)0, acc[nt], false, false);
    }
  }

  const int rbase = m0 + wave * 16 + ((lane >> 4) ? 8 : 0);
#pragma unroll
  for (int nt = 0; nt < 8; ++nt) {
    const int gcol = n0 + nt * 16 + (lane & 15);
    const float bv = bi[gcol];
#pragma unroll
    for (int v = 0; v < 8; ++v)
      xg[(size_t)(rbase + v) * G4 + gcol] = f2bf(acc[nt][v] + bv);
  }
}

// ---------------------------------------------------------------------------
// step(t): gates[64,4096] = xg_t + h @ Wh^T + bh, fused cell update.
// Block = hidden cols j0..j0+15 across all 4 gates; wave w -> gate g=w>>1,
// M-tile pair mp=w&1.  h double-buffered across launches; LDS tiles
// double-buffered against the TDM within a launch.
// ---------------------------------------------------------------------------
__global__ __launch_bounds__(256) void lstm_step(
    const unsigned short* __restrict__ xg, const unsigned short* __restrict__ whb,
    const float* __restrict__ bh, const unsigned short* __restrict__ hin,
    unsigned short* __restrict__ hnext, float* __restrict__ cbuf,
    float* __restrict__ out, int t) {
  __shared__ unsigned short sH[2][64 * 40];  // h tile, padded stride 40
  __shared__ unsigned short sW[2][64 * 40];  // Wh tile (4 gates x 16 rows)
  __shared__ unsigned short sXG[64 * 4 * 16];// xg_t slice: (b*4+g)*16+col
  __shared__ float sG[4 * 64 * 17];
  const int tid  = threadIdx.x;
  const int wave = tid >> 5, lane = tid & 31;
  const int j0 = blockIdx.x * 16;
  const int g  = wave >> 1;
  const int mp = wave & 1;
  f32x8 acc[2] = {};

#if USE_TDM
  const unsigned ldsH  = (unsigned)(size_t)(void*)&sH[0][0]; // [31:0]=LDS offset
  const unsigned ldsW  = (unsigned)(size_t)(void*)&sW[0][0];
  const unsigned ldsXG = (unsigned)(size_t)(void*)&sXG[0];
  if (wave == 0) {
    // xg_t slice once per step: 16 cols x 4 gates (stride 1024) x 64 batch
    // rows (stride S*4H), linear into sXG.
    tdm_load(ldsXG, xg + (size_t)t * G4 + j0, 16u, 4u, 64u,
             1024ull, (unsigned long long)Ssz * G4, 0);
    // chunk 0 tiles into buffer 0
    tdm_load(ldsH, hin, 32u, 64u, 0u, (unsigned long long)Hsz, 0ull, 1);
    tdm_load(ldsW, whb + (size_t)j0 * Hsz, 32u, 16u, 4u,
             (unsigned long long)Hsz, (unsigned long long)Hsz * Hsz, 1);
  }
#else
  {  // fallback: cooperative copy of the xg_t slice
    int e = tid;
    for (int i = 0; i < 2; ++i, e += 256) {
      int b = e >> 3, seg = e & 7;
      int gg = seg >> 1, hf = (seg & 1) * 8;
      U4 v = *(const U4*)(xg + ((size_t)b * Ssz + t) * G4 + gg * Hsz + j0 + hf);
      *(U4*)&sXG[(b * 4 + gg) * 16 + hf] = v;
    }
  }
#endif

  for (int it = 0; it < Hsz / 32; ++it) {
    const int k0 = it * 32;
    __syncthreads();   // barrier A: buffer (it+1)&1 fully read by compute it-1
#if USE_TDM
    if (wave == 0) {
      if (it + 1 < Hsz / 32) {
        const unsigned p = (unsigned)((it + 1) & 1) * (64u * 40u * 2u);
        tdm_load(ldsH + p, hin + (k0 + 32), 32u, 64u, 0u,
                 (unsigned long long)Hsz, 0ull, 1);
        tdm_load(ldsW + p, whb + (size_t)j0 * Hsz + (k0 + 32), 32u, 16u, 4u,
                 (unsigned long long)Hsz, (unsigned long long)Hsz * Hsz, 1);
        // TENSORcnt completes in-order: <=2 outstanding => chunk it landed.
        __builtin_amdgcn_s_wait_tensorcnt(2);
      } else {
        __builtin_amdgcn_s_wait_tensorcnt(0);
      }
    }
#else
    {
      int row = tid >> 2, c4 = tid & 3;
      *(U4*)&sH[it & 1][row * 40 + c4 * 8] =
          *(const U4*)(hin + (size_t)row * Hsz + k0 + c4 * 8);
      int n = (row >> 4) * Hsz + j0 + (row & 15);
      *(U4*)&sW[it & 1][row * 40 + c4 * 8] =
          *(const U4*)(whb + (size_t)n * Hsz + k0 + c4 * 8);
    }
#endif
    __syncthreads();   // barrier B: chunk it visible to all waves
    const unsigned short* cH = sH[it & 1];
    const unsigned short* cW = sW[it & 1];
    const int s     = (lane >> 4) * 8;
    const int khalf = (lane >> 4) * 16;
    Frag b;
    {
      const int brow = g * 16 + (lane & 15);
      b.q[0] = *(const U4*)&cW[brow * 40 + khalf];
      b.q[1] = *(const U4*)&cW[brow * 40 + khalf + 8];
    }
#pragma unroll
    for (int k = 0; k < 2; ++k) {
      const int arow = (mp * 2 + k) * 16 + (lane & 15);
      Frag a;
      a.q[0] = *(const U4*)&cH[arow * 40 + khalf];
      a.q[1] = *(const U4*)&cH[arow * 40 + khalf + 8];
      acc[k] = __builtin_amdgcn_wmma_f32_16x16x32_bf16(
          false, a.v, false, b.v, (short)0, acc[k], false, false);
    }
  }
  __syncthreads();

  {  // activations into LDS gate buffer (padded stride 17)
    const int col = lane & 15;
    const float bhv = bh[g * Hsz + j0 + col];
#pragma unroll
    for (int k = 0; k < 2; ++k) {
#pragma unroll
      for (int v = 0; v < 8; ++v) {
        const int brow = (mp * 2 + k) * 16 + v + ((lane >> 4) ? 8 : 0);
        float pre = acc[k][v] + bhv + bf2f(sXG[(brow * 4 + g) * 16 + col]);
        float act = (g == 2) ? tanhf(pre) : 1.0f / (1.0f + __expf(-pre));
        sG[(g * 64 + brow) * 17 + col] = act;
      }
    }
  }
  __syncthreads();

  for (int e = tid; e < 64 * 16; e += 256) {
    const int brow = e >> 4, col = e & 15;
    const int j = j0 + col;
    const float ig = sG[(brow) * 17 + col];
    const float fg = sG[(64 + brow) * 17 + col];
    const float gg = sG[(128 + brow) * 17 + col];
    const float og = sG[(192 + brow) * 17 + col];
    const size_t cj = (size_t)brow * Hsz + j;
    const float cold = cbuf[cj];
    const float cnew = fg * cold + ig * gg;
    const float hn = og * tanhf(cnew);
    cbuf[cj]  = cnew;
    hnext[cj] = f2bf(hn);
    out[((size_t)brow * Ssz + t) * Hsz + j] = hn;
    if (t == Ssz - 1) {                    // final (h, c) tuple outputs
      out[(size_t)Bsz * Ssz * Hsz + cj] = hn;
      out[(size_t)Bsz * Ssz * Hsz + (size_t)Bsz * Hsz + cj] = cnew;
    }
  }
}

// ---------------------------------------------------------------------------
extern "C" void kernel_launch(void* const* d_in, const int* in_sizes, int n_in,
                              void* d_out, int out_size, void* d_ws, size_t ws_size,
                              hipStream_t stream) {
  const float* x  = (const float*)d_in[0];
  const float* h0 = (const float*)d_in[1];
  const float* c0 = (const float*)d_in[2];
  const float* Wi = (const float*)d_in[3];
  const float* bi = (const float*)d_in[4];
  const float* Wh = (const float*)d_in[5];
  const float* bh = (const float*)d_in[6];
  float* out = (float*)d_out;

  // ws layout (re-initialized every call; ~265 MiB total):
  char* ws = (char*)d_ws;
  size_t off = 0;
  unsigned short* xg  = (unsigned short*)(ws + off); off += (size_t)Bsz * Ssz * G4 * 2;
  unsigned short* whb = (unsigned short*)(ws + off); off += (size_t)G4 * Hsz * 2;
  unsigned short* hb0 = (unsigned short*)(ws + off); off += (size_t)Bsz * Hsz * 2;
  unsigned short* hb1 = (unsigned short*)(ws + off); off += (size_t)Bsz * Hsz * 2;
  float*          cb  = (float*)(ws + off);          off += (size_t)Bsz * Hsz * 4;

  lstm_prep<<<dim3((G4 * Hsz + 255) / 256), 256, 0, stream>>>(Wh, h0, c0, whb, hb0, cb);
  lstm_proj_gemm<<<dim3((Bsz * Ssz) / 128, G4 / 128), 256, 0, stream>>>(x, Wi, bi, xg);
  for (int t = 0; t < Ssz; ++t) {
    const unsigned short* hin = (t & 1) ? hb1 : hb0;
    unsigned short* hnx       = (t & 1) ? hb0 : hb1;
    lstm_step<<<dim3(Hsz / 16), 256, 0, stream>>>(xg, whb, bh, hin, hnx, cb, out, t);
  }
}
